// RNNCell_80461917323399
// MI455X (gfx1250) — compile-verified, matrix operand order checked by reading, not run on hardware
//
#include <hip/hip_runtime.h>
#include <hip/hip_bf16.h>

typedef _Float16 v8h  __attribute__((ext_vector_type(8)));
typedef _Float16 v16h __attribute__((ext_vector_type(16)));
typedef float    v8f  __attribute__((ext_vector_type(8)));

#define B_    16
#define T_    384
#define I_    256
#define H_    256
#define G3_   768
#define KMAX_ 128

// ---- fast transcendentals (gfx1250 has V_TANH_F32 / V_EXP_F32 / V_RCP_F32) ----
#if __has_builtin(__builtin_amdgcn_tanhf)
__device__ __forceinline__ float fast_tanh(float x){ return __builtin_amdgcn_tanhf(x); }
#elif __has_builtin(__builtin_amdgcn_tanh_f32)
__device__ __forceinline__ float fast_tanh(float x){ return __builtin_amdgcn_tanh_f32(x); }
#else
__device__ __forceinline__ float fast_tanh(float x){ return tanhf(x); }
#endif
#if __has_builtin(__builtin_amdgcn_rcpf)
__device__ __forceinline__ float fast_rcp(float x){ return __builtin_amdgcn_rcpf(x); }
#else
__device__ __forceinline__ float fast_rcp(float x){ return 1.0f/x; }
#endif
__device__ __forceinline__ float sigmoidf_(float x){ return fast_rcp(1.0f + __expf(-x)); }

// ---- WMMA fragment helpers ----
// A-matrix 16x32 f16 layout (ISA 7.12.2): lane m=lane&15, g=lane>>4;
// halves j<8 -> K = 8g+j ; j>=8 -> K = 16+8g+(j-8). Two 16B loads per K=32.
// B uses the transpose-symmetric layout: lane n=lane&15 holds W[n][K...] of a
// row-major [N][K] weight (i.e. result = A @ W^T).
__device__ __forceinline__ v16h load_frag(const _Float16* rowp, int k0, int g){
  v16h f;
  ((v8h*)&f)[0] = *(const v8h*)(rowp + k0 + 8*g);
  ((v8h*)&f)[1] = *(const v8h*)(rowp + k0 + 16 + 8*g);
  return f;
}
__device__ __forceinline__ v8f wmma_f16(v16h a, v16h b, v8f c){
  return __builtin_amdgcn_wmma_f32_16x16x32_f16(false, a, false, b, (short)0, c, false, false);
}

// =====================================================================
// Kernel 0: convert weights/x to f16 in workspace, init P slot memory (f16).
// =====================================================================
__global__ void k_prep(const float* __restrict__ x, const float* __restrict__ hidden,
                       const float* __restrict__ w_ih, const float* __restrict__ w_hh,
                       const float* __restrict__ w_att, const float* __restrict__ w_inf,
                       const float* __restrict__ w_comb,
                       _Float16* __restrict__ x16, _Float16* __restrict__ wih16,
                       _Float16* __restrict__ whh16, _Float16* __restrict__ watt16,
                       _Float16* __restrict__ winf16, _Float16* __restrict__ wcomb16,
                       _Float16* __restrict__ P16)
{
  const int tid = blockIdx.x*blockDim.x + threadIdx.x;
  const int np  = gridDim.x*blockDim.x;
  for (int i=tid; i<B_*T_*I_; i+=np) x16[i]   = (_Float16)x[i];
  for (int i=tid; i<G3_*I_;  i+=np){ wih16[i] = (_Float16)w_ih[i];
                                     whh16[i] = (_Float16)w_hh[i]; }
  for (int i=tid; i<H_*H_;   i+=np){ watt16[i]= (_Float16)w_att[i];
                                     winf16[i]= (_Float16)w_inf[i]; }
  for (int i=tid; i<H_*2*H_; i+=np)  wcomb16[i]=(_Float16)w_comb[i];
  for (int i=tid; i<B_*KMAX_*H_; i+=np){
    const int h = i & (H_-1);
    const int k = (i >> 8) & (KMAX_-1);
    const int b = i >> 15;
    P16[i] = (k==0) ? (_Float16)hidden[b*H_ + h] : (_Float16)0.0f;
  }
}

// =====================================================================
// Kernel 1: GI = x @ w_ih^T + b_ih   ([6144 x 256] @ [256 x 768])
// grid: 384 blocks (one 16-row M tile each), 256 thr = 8 waves, 6 N-tiles/wave
// =====================================================================
__global__ void __launch_bounds__(256) k_gi(const _Float16* __restrict__ x16,
                                            const _Float16* __restrict__ wih16,
                                            const float* __restrict__ b_ih,
                                            float* __restrict__ GI)
{
  const int wave = threadIdx.x >> 5;
  const int lane = threadIdx.x & 31;
  const int g    = lane >> 4;
  const int mn   = lane & 15;
  const int m0   = blockIdx.x * 16;
  const _Float16* Arow = x16 + (m0 + mn)*I_;
  for (int nt = wave; nt < 48; nt += 8){
    const _Float16* Brow = wih16 + (nt*16 + mn)*I_;
    v8f c = {0.f,0.f,0.f,0.f,0.f,0.f,0.f,0.f};
    #pragma unroll
    for (int k0=0; k0<I_; k0+=32)
      c = wmma_f16(load_frag(Arow,k0,g), load_frag(Brow,k0,g), c);
    const int col  = nt*16 + mn;
    const float bi = b_ih[col];
    #pragma unroll
    for (int r=0; r<8; r++)
      GI[(m0 + r + 8*g)*G3_ + col] = c[r] + bi;
  }
}

// =====================================================================
// Kernel 2: sequential recurrence. 1 block, 512 thr = 16 wave32s.
// LDS: hA 8K + hB 16K + gh16 24K (proj aliased inside) + sc 8K + wm 1K = 57K
// =====================================================================
__global__ void __launch_bounds__(512) k_rnn(
    const _Float16* __restrict__ whh16,  const _Float16* __restrict__ watt16,
    const _Float16* __restrict__ winf16, const _Float16* __restrict__ wcomb16,
    const float* __restrict__ GI,        _Float16* __restrict__ P16,
    const float* __restrict__ b_hh,      const float* __restrict__ w_merge,
    const float* __restrict__ b_comb,    const float* __restrict__ hidden,
    float* __restrict__ out)
{
  __shared__ __align__(16) _Float16 hA[B_][H_];      // current h (f16)      8KB
  __shared__ __align__(16) _Float16 hB[B_][2*H_];    // [h_gru | ctx]       16KB
  __shared__ __align__(16) _Float16 gh16[B_*G3_];    // gates; proj alias   24KB
  __shared__ __align__(16) float    sc[B_][KMAX_];   // scores / attn        8KB
  __shared__              float     wm[H_];          // w_merge              1KB
  _Float16* proj = gh16;   // [B_][H_] alias: gh dead after gate phase

  const int tid  = threadIdx.x;
  const int wave = tid >> 5;
  const int lane = tid & 31;
  const int g    = lane >> 4;
  const int mn   = lane & 15;
  const int h0   = lane*8;            // per-thread element base within wave's batch
  const int bb   = wave;              // batch for elementwise phases

  // init hA = f16(hidden), wm = w_merge
  for (int i=tid; i<B_*H_; i+=512) hA[i>>8][i&255] = (_Float16)hidden[i];
  for (int i=tid; i<H_;    i+=512) wm[i] = w_merge[i];

  #pragma unroll 1
  for (int t=0; t<T_; t++){
    __syncthreads();
    // ---------- Phase 1: gh = h @ w_hh^T + b_hh (48 N-tiles, 3 per wave)
    #pragma unroll 1
    for (int it=0; it<3; ++it){
      const int nt = wave + 16*it;
      const _Float16* Brow = whh16 + (nt*16 + mn)*H_;
      const _Float16* Arow = &hA[mn][0];
      v8f c = {0.f,0.f,0.f,0.f,0.f,0.f,0.f,0.f};
      #pragma unroll
      for (int k0=0; k0<H_; k0+=32)
        c = wmma_f16(load_frag(Arow,k0,g), load_frag(Brow,k0,g), c);
      const int col  = nt*16 + mn;
      const float bh = b_hh[col];
      #pragma unroll
      for (int r=0; r<8; r++) gh16[(r + 8*g)*G3_ + col] = (_Float16)(c[r] + bh);
    }
    __syncthreads();
    // ---------- Phase 2: GRU gates -> h_gru into hB[:,0:256]
    {
      const float* gi = GI + (bb*T_ + t)*G3_ + h0;
      const _Float16* ghr = gh16 + bb*G3_ + h0;
      #pragma unroll
      for (int j=0; j<8; j++){
        const float rr = sigmoidf_(gi[j]       + (float)ghr[j]);
        const float zz = sigmoidf_(gi[256 + j] + (float)ghr[256 + j]);
        const float nn = fast_tanh(gi[512 + j] + rr*(float)ghr[512 + j]);
        const float hp = (float)hA[bb][h0 + j];
        hB[bb][h0 + j] = (_Float16)(nn + zz*(hp - nn));
      }
    }
    __syncthreads();
    // ---------- Phase 3: proj = h_gru @ w_att^T (16 tiles, 1 per wave)
    {
      const int nt = wave;
      const _Float16* Brow = watt16 + (nt*16 + mn)*H_;
      const _Float16* Arow = &hB[mn][0];
      v8f c = {0.f,0.f,0.f,0.f,0.f,0.f,0.f,0.f};
      #pragma unroll
      for (int k0=0; k0<H_; k0+=32)
        c = wmma_f16(load_frag(Arow,k0,g), load_frag(Brow,k0,g), c);
      const int col = nt*16 + mn;
      #pragma unroll
      for (int r=0; r<8; r++) proj[(r + 8*g)*H_ + col] = (_Float16)c[r];
    }
    __syncthreads();
    const int nvalid = t/3 + 1;   // slots k with 3k <= t
    // ---------- Phase 4: scores[b][k] = sum_h tanh(P+proj)*w_merge
    for (int k = wave; k < nvalid; k += 16){
      const _Float16* Pp = P16 + (mn*KMAX_ + k)*H_ + g*128;
      const _Float16* pj = proj + mn*H_ + g*128;
      float s = 0.f;
      #pragma unroll 2
      for (int j=0; j<128; j+=8){
        const v8h p = *(const v8h*)(Pp + j);
        #pragma unroll
        for (int u=0; u<8; u++)
          s += fast_tanh((float)p[u] + (float)pj[j+u]) * wm[g*128 + j + u];
      }
      s += __shfl_xor(s, 16, 32);
      if (g == 0) sc[mn][k] = s;
    }
    __syncthreads();
    // ---------- Phase 5: softmax over k (wave w owns batch w)
    {
      const int b = wave;
      float v0[4]; float mx = -1e30f;
      #pragma unroll
      for (int i=0;i<4;i++){
        const int k = lane + 32*i;
        v0[i] = (k < nvalid) ? sc[b][k] : -1e30f;
        mx = fmaxf(mx, v0[i]);
      }
      for (int off=16; off; off>>=1) mx = fmaxf(mx, __shfl_xor(mx, off, 32));
      float sum = 0.f; float e[4];
      #pragma unroll
      for (int i=0;i<4;i++){
        const int k = lane + 32*i;
        e[i] = (k < nvalid) ? __expf(v0[i] - mx) : 0.f;
        sum += e[i];
      }
      for (int off=16; off; off>>=1) sum += __shfl_xor(sum, off, 32);
      const float inv = fast_rcp(sum);
      #pragma unroll
      for (int i=0;i<4;i++){
        const int k = lane + 32*i;
        if (k < nvalid) sc[b][k] = e[i]*inv;
      }
    }
    __syncthreads();
    // ---------- Phase 6: ctx = attn . P  -> hB[:,256:512]
    {
      float acc[8] = {0.f,0.f,0.f,0.f,0.f,0.f,0.f,0.f};
      for (int k=0; k<nvalid; k++){
        const float a = sc[bb][k];
        const v8h p = *(const v8h*)(P16 + (bb*KMAX_ + k)*H_ + h0);
        #pragma unroll
        for (int u=0; u<8; u++) acc[u] += a*(float)p[u];
      }
      #pragma unroll
      for (int j=0;j<8;j++) hB[bb][H_ + h0 + j] = (_Float16)acc[j];
    }
    __syncthreads();
    // ---------- Phase 7: h = tanh([h_gru,ctx] @ w_comb^T + b_comb), K=512
    {
      const int nt = wave;
      const _Float16* Brow = wcomb16 + (nt*16 + mn)*(2*H_);
      const _Float16* Arow = &hB[mn][0];
      v8f c = {0.f,0.f,0.f,0.f,0.f,0.f,0.f,0.f};
      #pragma unroll
      for (int k0=0; k0<2*H_; k0+=32)
        c = wmma_f16(load_frag(Arow,k0,g), load_frag(Brow,k0,g), c);
      const int col  = nt*16 + mn;
      const float bc = b_comb[col];
      #pragma unroll
      for (int r=0; r<8; r++){
        const int m = r + 8*g;
        const float v = fast_tanh(c[r] + bc);
        out[(m*T_ + t)*H_ + col] = v;
        hA[m][col] = (_Float16)v;
      }
    }
    // ---------- Phase 8: every 3rd step, P[:,kw] = h @ w_inf^T (f16 store)
    if (((t+1) % 3) == 0 && ((t+1)/3) < KMAX_){
      __syncthreads();
      const int kw = (t+1)/3;
      const int nt = wave;
      const _Float16* Brow = winf16 + (nt*16 + mn)*H_;
      const _Float16* Arow = &hA[mn][0];
      v8f c = {0.f,0.f,0.f,0.f,0.f,0.f,0.f,0.f};
      #pragma unroll
      for (int k0=0; k0<H_; k0+=32)
        c = wmma_f16(load_frag(Arow,k0,g), load_frag(Brow,k0,g), c);
      const int col = nt*16 + mn;
      #pragma unroll
      for (int r=0; r<8; r++){
        const int m = r + 8*g;
        P16[(m*KMAX_ + kw)*H_ + col] = (_Float16)c[r];
      }
      __threadfence();   // make P visible to next step's loads
    }
  }
}

// =====================================================================
// Workspace layout (bytes, all offsets 256B-aligned)
// =====================================================================
#define OFF_X16     0u                   // 16*384*256 f16 = 3,145,728
#define OFF_WIH16   3145728u             // 768*256 f16    =   393,216
#define OFF_WHH16   3538944u             // 768*256 f16    =   393,216
#define OFF_WATT16  3932160u             // 256*256 f16    =   131,072
#define OFF_WINF16  4063232u             // 256*256 f16    =   131,072
#define OFF_WCOMB16 4194304u             // 256*512 f16    =   262,144
#define OFF_GI      4456448u             // 6144*768 f32   = 18,874,368
#define OFF_P       23330816u            // 16*128*256 f16 =  1,048,576
// total ~24.4 MB

extern "C" void kernel_launch(void* const* d_in, const int* in_sizes, int n_in,
                              void* d_out, int out_size, void* d_ws, size_t ws_size,
                              hipStream_t stream) {
  const float* x       = (const float*)d_in[0];
  const float* hidden  = (const float*)d_in[1];
  const float* w_ih    = (const float*)d_in[2];
  const float* w_hh    = (const float*)d_in[3];
  const float* b_ih    = (const float*)d_in[4];
  const float* b_hh    = (const float*)d_in[5];
  const float* w_att   = (const float*)d_in[6];
  const float* w_inf   = (const float*)d_in[7];
  const float* w_merge = (const float*)d_in[8];
  const float* w_comb  = (const float*)d_in[9];
  const float* b_comb  = (const float*)d_in[10];
  float* out = (float*)d_out;

  char* ws = (char*)d_ws;
  _Float16* x16     = (_Float16*)(ws + OFF_X16);
  _Float16* wih16   = (_Float16*)(ws + OFF_WIH16);
  _Float16* whh16   = (_Float16*)(ws + OFF_WHH16);
  _Float16* watt16  = (_Float16*)(ws + OFF_WATT16);
  _Float16* winf16  = (_Float16*)(ws + OFF_WINF16);
  _Float16* wcomb16 = (_Float16*)(ws + OFF_WCOMB16);
  float*    GI      = (float*)(ws + OFF_GI);
  _Float16* P16     = (_Float16*)(ws + OFF_P);

  k_prep<<<512, 256, 0, stream>>>(x, hidden, w_ih, w_hh, w_att, w_inf, w_comb,
                                  x16, wih16, whh16, watt16, winf16, wcomb16, P16);
  k_gi<<<(B_*T_)/16, 256, 0, stream>>>(x16, wih16, b_ih, GI);
  k_rnn<<<1, 512, 0, stream>>>(whh16, watt16, winf16, wcomb16, GI, P16,
                               b_hh, w_merge, b_comb, hidden, out);
}
